// Attention_11304353923883
// MI455X (gfx1250) — matrix operand order repaired and checked by reading, not compiled, not run on hardware
//
#include <hip/hip_runtime.h>

// ---------------------------------------------------------------------------
// MHA forward for MI455X (gfx1250, wave32).
//   * all matmuls via v_wmma_f32_16x16x32_bf16 (bf16 in, fp32 accumulate)
//   * attention K/V tiles staged block-wide into LDS by the Tensor Data
//     Mover (tensor_load_to_lds + s_wait_tensorcnt), double-buffered
//   * GEMM waves compute 32x64 tiles: 2 A-frags amortize 4 B-frags (8 WMMA/K-step)
// Workspace (bf16, ~48 MiB): Xb[4096][1024], WqkvT[3072][1024],
//   WoutT[1024][1024], Q[2][16][2048][64] (pre-scaled), K[2][16][2048][64],
//   Vt[2][16][64][2048], O[4096][1024]
// ---------------------------------------------------------------------------

typedef __bf16 bf16_t;
typedef __attribute__((ext_vector_type(16))) __bf16 v16bf;
typedef __attribute__((ext_vector_type(8)))  __bf16 v8bf;
typedef __attribute__((ext_vector_type(8)))  float  v8f;
typedef __attribute__((ext_vector_type(4)))  unsigned int v4u;
typedef __attribute__((ext_vector_type(8)))  int v8i;
typedef __attribute__((ext_vector_type(4)))  int v4i;

union AFrag { v16bf v; v8bf h[2]; };

#define SEQ   2048
#define DIM   1024
#define NQKV  3072
#define NROWS 4096   // b * n

#if defined(__has_builtin)
#if __has_builtin(__builtin_amdgcn_tensor_load_to_lds)
#define HAVE_TDM 1
#endif
#endif
#ifndef HAVE_TDM
#define HAVE_TDM 0
#endif

__device__ __forceinline__ v8f wmma_bf16(v16bf a, v16bf b, v8f c) {
    // (neg_a, A, neg_b, B, c_mod, C, reuse_a, reuse_b)
    return __builtin_amdgcn_wmma_f32_16x16x32_bf16(false, a, false, b,
                                                   (short)0, c, false, false);
}
__device__ __forceinline__ v16bf ldv16(const bf16_t* p) { return *(const v16bf*)p; }
__device__ __forceinline__ v8bf  ldv8 (const bf16_t* p) { return *(const v8bf*)p; }
__device__ __forceinline__ v8f   vzero8() { v8f z = {0.f,0.f,0.f,0.f,0.f,0.f,0.f,0.f}; return z; }

#if HAVE_TDM
// Issue a 2D TDM tile load: tile_d1 rows x tile_d0 elems (bf16), row stride
// stride_d0 (elems) in global, packed contiguously into LDS at lds_off.
__device__ __forceinline__ void tdm_load_2d(unsigned lds_off, const bf16_t* gaddr,
                                            unsigned tile_d0, unsigned tile_d1,
                                            unsigned long long stride_d0,
                                            unsigned tensor_d0, unsigned tensor_d1) {
    unsigned long long ga = (unsigned long long)(uintptr_t)gaddr;
    v4u g0 = {0u, 0u, 0u, 0u};
    g0[0] = 1u;                                     // count=1, user descriptor
    g0[1] = lds_off;                                // lds_addr (bytes)
    g0[2] = (unsigned)(ga & 0xffffffffu);           // global_addr[31:0]
    g0[3] = (unsigned)((ga >> 32) & 0x01ffffffu) | (2u << 30);  // addr[56:32] | type=2
    v8i g1 = {0,0,0,0,0,0,0,0};
    g1[0] = (int)(1u << 16);                        // wg_mask=0, data_size=1 (2 bytes)
    g1[1] = (int)((tensor_d0 & 0xffffu) << 16);     // tensor_dim0[15:0]
    g1[2] = (int)(((tensor_d0 >> 16) & 0xffffu) | ((tensor_d1 & 0xffffu) << 16));
    g1[3] = (int)(((tensor_d1 >> 16) & 0xffffu) | ((tile_d0 & 0xffffu) << 16));
    g1[4] = (int)(tile_d1 & 0xffffu);               // tile_dim1 (tile_dim2=0 -> 2D)
    g1[5] = (int)(unsigned)(stride_d0 & 0xffffffffu);   // tensor_dim0_stride[31:0]
    g1[6] = (int)(unsigned)((stride_d0 >> 32) & 0xffffu);
    g1[7] = 0;
    v4i z4 = {0, 0, 0, 0};
#if defined(__clang_major__) && (__clang_major__ >= 23)
    v8i z8 = {0,0,0,0,0,0,0,0};
    __builtin_amdgcn_tensor_load_to_lds(g0, g1, z4, z4, z8, 0);
#else
    __builtin_amdgcn_tensor_load_to_lds(g0, g1, z4, z4, 0);
#endif
}
#endif

// ---------------------------------------------------------------- prep ----
__global__ void cvt_f32_bf16(const float* __restrict__ in, bf16_t* __restrict__ out, int n) {
    int i = blockIdx.x * blockDim.x + threadIdx.x;
    if (i < n) out[i] = (bf16_t)in[i];
}
__global__ void transpose_f32_bf16(const float* __restrict__ in, bf16_t* __restrict__ out,
                                   int rows, int cols) {
    int i = blockIdx.x * blockDim.x + threadIdx.x;
    if (i >= rows * cols) return;
    int c = i % cols, r = i / cols;
    out[(size_t)c * rows + r] = (bf16_t)in[(size_t)r * cols + c];
}

// ---------------------------------------------------- QKV projection ------
// C[4096,3072] = Xb * Wqkv; each wave: 32 rows x 64 cols (8 WMMA / K-step).
__global__ __launch_bounds__(128) void qkv_gemm(const bf16_t* __restrict__ Xb,
                                                const bf16_t* __restrict__ WT,
                                                bf16_t* __restrict__ Qp,
                                                bf16_t* __restrict__ Kp,
                                                bf16_t* __restrict__ Vtp) {
    const int lane = threadIdx.x & 31;
    const int wid  = threadIdx.x >> 5;
    const int gw   = blockIdx.x * 4 + wid;
    const int TN   = NQKV / 64;                    // 48 col tiles
    const int tn = gw % TN, tm = gw / TN;          // tm in [0,128)
    const int m0 = tm * 32, n0 = tn * 64;
    const int hi = lane >> 4, l15 = lane & 15;

    v8f acc[2][4];
#pragma unroll
    for (int g = 0; g < 2; ++g)
#pragma unroll
        for (int t = 0; t < 4; ++t) acc[g][t] = vzero8();

    const bf16_t* arow0 = Xb + (size_t)(m0 + l15) * DIM;
    const bf16_t* arow1 = Xb + (size_t)(m0 + 16 + l15) * DIM;
    for (int k0 = 0; k0 < DIM; k0 += 32) {
        AFrag a0, a1;
        a0.h[0] = ldv8(arow0 + k0 + hi * 16);
        a0.h[1] = ldv8(arow0 + k0 + 8 + hi * 16);
        a1.h[0] = ldv8(arow1 + k0 + hi * 16);
        a1.h[1] = ldv8(arow1 + k0 + 8 + hi * 16);
#pragma unroll
        for (int t = 0; t < 4; ++t) {
            v16bf b = ldv16(WT + (size_t)(n0 + t * 16 + l15) * DIM + k0 + hi * 16);
            acc[0][t] = wmma_bf16(a0.v, b, acc[0][t]);
            acc[1][t] = wmma_bf16(a1.v, b, acc[1][t]);
        }
    }
#pragma unroll
    for (int t = 0; t < 4; ++t) {
        int col = n0 + t * 16 + l15;
        int which = col >> 10;            // 0=q 1=k 2=v (wave-uniform)
        int h = (col >> 6) & 15;
        int d = col & 63;
#pragma unroll
        for (int g = 0; g < 2; ++g)
#pragma unroll
            for (int r = 0; r < 8; ++r) {
                int row = m0 + g * 16 + r + hi * 8;
                int b = row >> 11, n = row & (SEQ - 1);
                size_t bh = (size_t)(b * 16 + h);
                float v = acc[g][t][r];
                if (which == 0)      Qp [(bh * SEQ + n) * 64 + d] = (bf16_t)(v * 0.125f);
                else if (which == 1) Kp [(bh * SEQ + n) * 64 + d] = (bf16_t)v;
                else                 Vtp[(bh * 64 + d) * SEQ + n] = (bf16_t)v;
            }
    }
}

// ---------------------------------------------------- flash attention -----
// Block = 4 waves sharing one (b,h); 64 query rows per block, 16 per wave.
// K/V tiles (32 keys) staged into LDS by TDM, double-buffered.
__global__ __launch_bounds__(128) void attn_fwd(const bf16_t* __restrict__ Qp,
                                                const bf16_t* __restrict__ Kp,
                                                const bf16_t* __restrict__ Vtp,
                                                bf16_t* __restrict__ Op) {
    __shared__ __align__(128) bf16_t kbuf[2][32 * 64];  // [key][d] packed
    __shared__ __align__(128) bf16_t vbuf[2][64 * 32];  // [d][key] packed
    __shared__ __align__(64)  bf16_t pl[4][16 * 48];    // per-wave P staging

    const int lane = threadIdx.x & 31;
    const int wid  = threadIdx.x >> 5;
    const int gw   = blockIdx.x * 4 + wid;
    const int qt = gw & 127;
    const int bh = gw >> 7;
    const int hi = lane >> 4, l15 = lane & 15;
    const int m0 = qt * 16;

    const bf16_t* Qb = Qp + (size_t)bh * SEQ * 64;
    const bf16_t* Kb = Kp + (size_t)bh * SEQ * 64;
    const bf16_t* Vb = Vtp + (size_t)bh * 64 * SEQ;

#if HAVE_TDM
    const unsigned koff0 = (unsigned)(uintptr_t)&kbuf[0][0];
    const unsigned koff1 = (unsigned)(uintptr_t)&kbuf[1][0];
    const unsigned voff0 = (unsigned)(uintptr_t)&vbuf[0][0];
    const unsigned voff1 = (unsigned)(uintptr_t)&vbuf[1][0];
    // prologue: stage tile 0 into buffer 0 (wave 0 only; TENSORcnt tracked there)
    if (wid == 0) {
        tdm_load_2d(koff0, Kb, 64, 32, 64, 64, SEQ);           // K rows contiguous
        tdm_load_2d(voff0, Vb, 32, 64, SEQ, SEQ, 64);          // Vt rows strided
    }
#else
    {   // cooperative copy fallback: tile 0 -> buffer 0
        const int tid = threadIdx.x;
#pragma unroll
        for (int s = 0; s < 2; ++s)
            *(v8bf*)&kbuf[0][(s * 128 + tid) * 8] = ldv8(Kb + (size_t)(s * 128 + tid) * 8);
#pragma unroll
        for (int s = 0; s < 2; ++s) {
            int idx = s * 128 + tid, row = idx >> 2, ch = idx & 3;
            *(v8bf*)&vbuf[0][row * 32 + ch * 8] = ldv8(Vb + (size_t)row * SEQ + ch * 8);
        }
    }
#endif

    // Q A-fragments for head-dim chunks [0,32) and [32,64), kept in VGPRs.
    AFrag qa0, qa1;
    {
        const bf16_t* qrow = Qb + (size_t)(m0 + l15) * 64;
        qa0.h[0] = ldv8(qrow + hi * 16);
        qa0.h[1] = ldv8(qrow + 8 + hi * 16);
        qa1.h[0] = ldv8(qrow + 32 + hi * 16);
        qa1.h[1] = ldv8(qrow + 40 + hi * 16);
    }

    v8f o[4]; o[0] = vzero8(); o[1] = vzero8(); o[2] = vzero8(); o[3] = vzero8();
    float mrun[8], lrun[8];
#pragma unroll
    for (int r = 0; r < 8; ++r) { mrun[r] = -3.0e38f; lrun[r] = 0.f; }

    bf16_t* P = &pl[wid][0];

    for (int j0 = 0; j0 < SEQ; j0 += 32) {
        const int cur = (j0 >> 5) & 1, nxt = cur ^ 1;
        const bool last = (j0 + 32 >= SEQ);
#if HAVE_TDM
        if (wid == 0) {
            if (!last) {   // overlap next tile's DMA with this tile's compute
                tdm_load_2d(nxt ? koff1 : koff0, Kb + (size_t)(j0 + 32) * 64, 64, 32, 64, 64, SEQ);
                tdm_load_2d(nxt ? voff1 : voff0, Vb + (j0 + 32), 32, 64, SEQ, SEQ, 64);
                __builtin_amdgcn_s_wait_tensorcnt((short)2);   // current tile complete
            } else {
                __builtin_amdgcn_s_wait_tensorcnt((short)0);
            }
        }
#else
        if (!last) {
            const int tid = threadIdx.x;
#pragma unroll
            for (int s = 0; s < 2; ++s)
                *(v8bf*)&kbuf[nxt][(s * 128 + tid) * 8] =
                    ldv8(Kb + (size_t)(j0 + 32) * 64 + (size_t)(s * 128 + tid) * 8);
#pragma unroll
            for (int s = 0; s < 2; ++s) {
                int idx = s * 128 + tid, row = idx >> 2, ch = idx & 3;
                *(v8bf*)&vbuf[nxt][row * 32 + ch * 8] =
                    ldv8(Vb + (size_t)row * SEQ + (j0 + 32) + ch * 8);
            }
        }
#endif
        __syncthreads();   // publish buffer `cur` to all waves

        const bf16_t* kl = &kbuf[cur][0];
        const bf16_t* vl = &vbuf[cur][0];

        // ---- S = Q K^T for 32 keys -> two 16x16 f32 tiles (B from LDS) ----
        v8f s0 = vzero8(), s1 = vzero8();
        {
            v16bf b00 = ldv16(kl + (l15) * 64 + hi * 16);
            v16bf b01 = ldv16(kl + (l15) * 64 + 32 + hi * 16);
            v16bf b10 = ldv16(kl + (16 + l15) * 64 + hi * 16);
            v16bf b11 = ldv16(kl + (16 + l15) * 64 + 32 + hi * 16);
            s0 = wmma_bf16(qa0.v, b00, s0);
            s0 = wmma_bf16(qa1.v, b01, s0);
            s1 = wmma_bf16(qa0.v, b10, s1);
            s1 = wmma_bf16(qa1.v, b11, s1);
        }

        // ---- online softmax (fp32); P to per-wave LDS as bf16 -------------
#pragma unroll
        for (int r = 0; r < 8; ++r) {
            float a0 = s0[r], a1 = s1[r];
            float mx = fmaxf(a0, a1);
#pragma unroll
            for (int m = 1; m < 16; m <<= 1) mx = fmaxf(mx, __shfl_xor(mx, m, 32));
            float mn   = fmaxf(mrun[r], mx);
            float corr = __expf(mrun[r] - mn);
            float p0 = __expf(a0 - mn), p1 = __expf(a1 - mn);
            float rs = p0 + p1;
#pragma unroll
            for (int m = 1; m < 16; m <<= 1) rs += __shfl_xor(rs, m, 32);
            lrun[r] = lrun[r] * corr + rs;
            mrun[r] = mn;
#pragma unroll
            for (int t = 0; t < 4; ++t) o[t][r] *= corr;
            int row = r + hi * 8;
            P[row * 48 + l15]      = (bf16_t)p0;
            P[row * 48 + 16 + l15] = (bf16_t)p1;
        }

        // ---- O += P V  (A = P from LDS, B = V tile from LDS) --------------
        AFrag pa;
        pa.h[0] = ldv8(P + l15 * 48 + hi * 16);
        pa.h[1] = ldv8(P + l15 * 48 + 8 + hi * 16);
#pragma unroll
        for (int t = 0; t < 4; ++t) {
            v16bf vbfr = ldv16(vl + (t * 16 + l15) * 32 + hi * 16);
            o[t] = wmma_bf16(pa.v, vbfr, o[t]);
        }
        __syncthreads();   // everyone done reading buffer `cur`
    }

    // ---- normalize, write O[b][n][h*64+d] as bf16 -------------------------
    const int b = bh >> 4, h = bh & 15;
#pragma unroll
    for (int r = 0; r < 8; ++r) {
        float inv = 1.f / lrun[r];
        int row = m0 + r + hi * 8;
        bf16_t* orow = Op + ((size_t)(b * SEQ + row)) * DIM + h * 64;
#pragma unroll
        for (int t = 0; t < 4; ++t) orow[t * 16 + l15] = (bf16_t)(o[t][r] * inv);
    }
}

// ---------------------------------------------------- output projection ---
__global__ __launch_bounds__(128) void out_gemm(const bf16_t* __restrict__ Ob,
                                                const bf16_t* __restrict__ WT,
                                                float* __restrict__ out) {
    const int lane = threadIdx.x & 31;
    const int wid  = threadIdx.x >> 5;
    const int gw   = blockIdx.x * 4 + wid;
    const int TN   = DIM / 64;                    // 16 col tiles
    const int tn = gw % TN, tm = gw / TN;         // tm in [0,128)
    const int m0 = tm * 32, n0 = tn * 64;
    const int hi = lane >> 4, l15 = lane & 15;

    v8f acc[2][4];
#pragma unroll
    for (int g = 0; g < 2; ++g)
#pragma unroll
        for (int t = 0; t < 4; ++t) acc[g][t] = vzero8();

    const bf16_t* arow0 = Ob + (size_t)(m0 + l15) * DIM;
    const bf16_t* arow1 = Ob + (size_t)(m0 + 16 + l15) * DIM;
    for (int k0 = 0; k0 < DIM; k0 += 32) {
        AFrag a0, a1;
        a0.h[0] = ldv8(arow0 + k0 + hi * 16);
        a0.h[1] = ldv8(arow0 + k0 + 8 + hi * 16);
        a1.h[0] = ldv8(arow1 + k0 + hi * 16);
        a1.h[1] = ldv8(arow1 + k0 + 8 + hi * 16);
#pragma unroll
        for (int t = 0; t < 4; ++t) {
            v16bf b = ldv16(WT + (size_t)(n0 + t * 16 + l15) * DIM + k0 + hi * 16);
            acc[0][t] = wmma_bf16(a0.v, b, acc[0][t]);
            acc[1][t] = wmma_bf16(a1.v, b, acc[1][t]);
        }
    }
#pragma unroll
    for (int g = 0; g < 2; ++g)
#pragma unroll
        for (int t = 0; t < 4; ++t)
#pragma unroll
            for (int r = 0; r < 8; ++r) {
                int row = m0 + g * 16 + r + hi * 8;
                out[(size_t)row * DIM + n0 + t * 16 + l15] = acc[g][t][r];
            }
}

// ---------------------------------------------------------------------------
extern "C" void kernel_launch(void* const* d_in, const int* in_sizes, int n_in,
                              void* d_out, int out_size, void* d_ws, size_t ws_size,
                              hipStream_t stream) {
    const float* x     = (const float*)d_in[0];   // [2,2048,1024]
    const float* w_qkv = (const float*)d_in[1];   // [1024,3072]
    const float* w_out = (const float*)d_in[2];   // [1024,1024]
    float* out = (float*)d_out;

    char* ws = (char*)d_ws;
    size_t off = 0;
    auto carve = [&](size_t bytes) { void* p = ws + off; off += (bytes + 255) & ~(size_t)255; return p; };
    bf16_t* Xb    = (bf16_t*)carve((size_t)NROWS * DIM * 2);
    bf16_t* WqkvT = (bf16_t*)carve((size_t)NQKV * DIM * 2);
    bf16_t* WoutT = (bf16_t*)carve((size_t)DIM * DIM * 2);
    bf16_t* Qp    = (bf16_t*)carve((size_t)NROWS * DIM * 2);
    bf16_t* Kp    = (bf16_t*)carve((size_t)NROWS * DIM * 2);
    bf16_t* Vtp   = (bf16_t*)carve((size_t)NROWS * DIM * 2);
    bf16_t* Ob    = (bf16_t*)carve((size_t)NROWS * DIM * 2);

    {
        int n = NROWS * DIM;
        cvt_f32_bf16<<<(n + 255) / 256, 256, 0, stream>>>(x, Xb, n);
        int nq = DIM * NQKV;
        transpose_f32_bf16<<<(nq + 255) / 256, 256, 0, stream>>>(w_qkv, WqkvT, DIM, NQKV);
        int no = DIM * DIM;
        transpose_f32_bf16<<<(no + 255) / 256, 256, 0, stream>>>(w_out, WoutT, DIM, DIM);
    }
    // QKV projection: 128 row-tiles(32) * 48 col-tiles, 4 waves/block
    qkv_gemm<<<(128 * 48) / 4, 128, 0, stream>>>(Xb, WqkvT, Qp, Kp, Vtp);
    // attention: 32 (b,h) * 128 query tiles, 4 waves/block
    attn_fwd<<<(32 * 128) / 4, 128, 0, stream>>>(Qp, Kp, Vtp, Ob);
    // output projection: 128 row-tiles(32) * 16 col-tiles, 4 waves/block
    out_gemm<<<(128 * 16) / 4, 128, 0, stream>>>(Ob, WoutT, out);
}
